// EfficientNetGAT_12343736009101
// MI455X (gfx1250) — compile-verified, matrix operand order checked by reading, not run on hardware
//
#include <hip/hip_runtime.h>
#include <hip/hip_bf16.h>
#include <cstdint>

// ---------------------------------------------------------------------------
// EfficientNet-GAT forward, MI455X (gfx1250), wave32.
// WMMA f16 (f32 accum) for all GEMMs; async Global->LDS double buffering.
// ---------------------------------------------------------------------------

#define N_NODES   49
#define FEAT_DIM  1280
#define NODE_DIM  26
#define KPAD      32      // NODE_DIM padded to WMMA K
#define MPAD      64      // N_NODES padded to 4x16 rows
#define GAT_HID   256
#define HEADS     4
#define NUM_CLS   5
#define BATCH     8192
#define N_TR      1274    // 49*26
#define NPAD_TR   1280    // padded N for transform GEMM
#define E_CNT     361

typedef __attribute__((ext_vector_type(16))) _Float16 v16h;
typedef __attribute__((ext_vector_type(8)))  float    v8f;

union FragU { v16h v; uint4 q[2]; };

__device__ __forceinline__ v8f wmma16(v16h a, v16h b, v8f c) {
  // (neg_a, A, neg_b, B, c_mod, C, reuse_a, reuse_b)
  return __builtin_amdgcn_wmma_f32_16x16x32_f16(false, a, false, b, (short)0, c,
                                                false, false);
}

// Async Global -> LDS 16-byte copy (per lane), tracked by ASYNCcnt.
// Inline asm: portable across ROCm 7.2 / amdgpu-toolchain builtin arity skew.
__device__ __forceinline__ void async_g2l_b128(unsigned lds_off, const void* gptr) {
  asm volatile("global_load_async_to_lds_b128 %0, %1, off"
               :: "v"(lds_off), "v"(gptr) : "memory");
}
__device__ __forceinline__ void wait_async0() {
  asm volatile("s_wait_asynccnt 0x0" ::: "memory");
}

// A-fragment (16x32 f16) from a row-major f16 matrix (also B-fragment when the
// matrix is stored transposed: lane holds one column of B = one row of B^T).
// Lane l<16: row l, K {0..7, 16..23}; lane>=16: row l-16, K {8..15, 24..31}.
__device__ __forceinline__ v16h gfrag(const _Float16* base, int row0, int ld, int k0) {
  int lane = threadIdx.x & 31;
  int r = row0 + (lane & 15);
  int koff = k0 + ((lane < 16) ? 0 : 8);
  FragU f;
  const _Float16* p = base + (size_t)r * ld + koff;
  f.q[0] = *reinterpret_cast<const uint4*>(p);
  f.q[1] = *reinterpret_cast<const uint4*>(p + 16);
  return f.v;
}

__device__ __forceinline__ v16h sfrag(const _Float16* s, int row0, int ld, int k0) {
  int lane = threadIdx.x & 31;
  int r = row0 + (lane & 15);
  int koff = k0 + ((lane < 16) ? 0 : 8);
  FragU f;
  f.q[0] = *reinterpret_cast<const uint4*>(s + r * ld + koff);
  f.q[1] = *reinterpret_cast<const uint4*>(s + r * ld + koff + 16);
  return f.v;
}

// ---------------------------------------------------------------------------
// K0: packing / conversion kernels
// ---------------------------------------------------------------------------
__global__ void k_zero_f16(_Float16* p, size_t n8) {  // n8 = count of uint4
  size_t i = (size_t)blockIdx.x * blockDim.x + threadIdx.x;
  if (i < n8) reinterpret_cast<uint4*>(p)[i] = make_uint4(0u, 0u, 0u, 0u);
}

__global__ void k_cvt_f16(const float* src, _Float16* dst, size_t n) {
  size_t i = (size_t)blockIdx.x * blockDim.x + threadIdx.x;
  if (i < n) dst[i] = (_Float16)src[i];
}

// wtT[n][k] = n < 1274 ? W[k][n] : 0 ; W row-major [1280][1274]
__global__ void k_pack_wt(const float* W, _Float16* wtT) {
  size_t i = (size_t)blockIdx.x * blockDim.x + threadIdx.x;
  if (i >= (size_t)NPAD_TR * FEAT_DIM) return;
  int n = (int)(i / FEAT_DIM), k = (int)(i % FEAT_DIM);
  wtT[i] = (n < N_TR) ? (_Float16)W[(size_t)k * N_TR + n] : (_Float16)0.f;
}

// g1 weights: wlT/wrT [256][32], K padded 26->32
__global__ void k_pack_g1(const float* wl, const float* wr, _Float16* wlT, _Float16* wrT) {
  int i = blockIdx.x * blockDim.x + threadIdx.x;
  if (i >= GAT_HID * KPAD) return;
  int n = i >> 5, k = i & 31;
  float a = (k < NODE_DIM) ? wl[k * GAT_HID + n] : 0.f;
  float b = (k < NODE_DIM) ? wr[k * GAT_HID + n] : 0.f;
  wlT[i] = (_Float16)a;
  wrT[i] = (_Float16)b;
}

// g2 weights: wlT2/wrT2 [256][256] transposed
__global__ void k_pack_g2(const float* wl, const float* wr, _Float16* wlT2, _Float16* wrT2) {
  int i = blockIdx.x * blockDim.x + threadIdx.x;
  if (i >= GAT_HID * GAT_HID) return;
  int n = i >> 8, k = i & 255;
  wlT2[i] = (_Float16)wl[k * GAT_HID + n];
  wrT2[i] = (_Float16)wr[k * GAT_HID + n];
}

// dst-sorted CSR for the static grid graph (built once per launch, serially).
__global__ void k_build_csr(const int* ei, int* esrc, int* edst, int* coff, int* ceid) {
  if (threadIdx.x != 0 || blockIdx.x != 0) return;
  int cnt[N_NODES];
  for (int n = 0; n < N_NODES; n++) cnt[n] = 0;
  for (int e = 0; e < E_CNT; e++) {
    int s = ei[e], d = ei[E_CNT + e];
    esrc[e] = s; edst[e] = d;
    cnt[d]++;
  }
  coff[0] = 0;
  for (int n = 0; n < N_NODES; n++) coff[n + 1] = coff[n] + cnt[n];
  int cur[N_NODES];
  for (int n = 0; n < N_NODES; n++) cur[n] = coff[n];
  for (int e = 0; e < E_CNT; e++) { int d = edst[e]; ceid[cur[d]++] = e; }
}

// ---------------------------------------------------------------------------
// K1: transform GEMM  node = x @ W + b   (M=8192, N=1274->1280, K=1280)
// 128x64 tile / WG (8 waves, each 32x32 = 2x2 WMMA).
// Double-buffered async Global->LDS staging (ASYNCcnt), overlap fill/compute.
// Output scattered into zero-padded node_pad [B][64][32] f16.
// ---------------------------------------------------------------------------
__global__ __launch_bounds__(256) void k_transform(const _Float16* __restrict__ xb,
                                                   const _Float16* __restrict__ wtT,
                                                   const float* __restrict__ tb,
                                                   _Float16* __restrict__ node_pad) {
  __shared__ _Float16 sA[2][128 * 32];  // 2 x 8 KB
  __shared__ _Float16 sB[2][64 * 32];   // 2 x 4 KB
  const int n0 = blockIdx.x * 64;
  const int m0 = blockIdx.y * 128;
  const int tid = threadIdx.x;
  const int wave = tid >> 5, lane = tid & 31;
  const int wm = wave >> 1, wn = wave & 1;

  auto stage = [&](int k0, int buf) {
    // A: 128x32 halves = 512 x 16B chunks (2 per thread)
    for (int u = tid; u < 512; u += 256) {
      int row = u >> 2, c8 = (u & 3) * 8;
      async_g2l_b128((unsigned)(uintptr_t)&sA[buf][row * 32 + c8],
                     &xb[(size_t)(m0 + row) * FEAT_DIM + k0 + c8]);
    }
    // B: 64x32 halves = 256 x 16B chunks (1 per thread)
    {
      int row = tid >> 2, c8 = (tid & 3) * 8;
      async_g2l_b128((unsigned)(uintptr_t)&sB[buf][row * 32 + c8],
                     &wtT[(size_t)(n0 + row) * FEAT_DIM + k0 + c8]);
    }
  };

  v8f acc[2][2] = {};
  stage(0, 0);
  wait_async0();
  __syncthreads();
  int cur = 0;
  for (int k0 = 0; k0 < FEAT_DIM; k0 += 32) {
    if (k0 + 32 < FEAT_DIM) stage(k0 + 32, cur ^ 1);  // overlap next fill
    v16h af0 = sfrag(sA[cur], wm * 32, 32, 0);
    v16h af1 = sfrag(sA[cur], wm * 32 + 16, 32, 0);
    v16h bf0 = sfrag(sB[cur], wn * 32, 32, 0);
    v16h bf1 = sfrag(sB[cur], wn * 32 + 16, 32, 0);
    acc[0][0] = wmma16(af0, bf0, acc[0][0]);
    acc[0][1] = wmma16(af0, bf1, acc[0][1]);
    acc[1][0] = wmma16(af1, bf0, acc[1][0]);
    acc[1][1] = wmma16(af1, bf1, acc[1][1]);
    wait_async0();
    __syncthreads();
    cur ^= 1;
  }
  // epilogue: scatter to node_pad[b][row49][ch26]
  for (int i = 0; i < 2; i++)
    for (int j = 0; j < 2; j++) {
      int col = n0 + wn * 32 + j * 16 + (lane & 15);
      if (col < N_TR) {
        float bias = tb[col];
        int nr = col / NODE_DIM;
        int nc = col - nr * NODE_DIM;
        for (int r = 0; r < 8; r++) {
          int m = m0 + wm * 32 + i * 16 + r + ((lane < 16) ? 0 : 8);
          float v = acc[i][j][r] + bias;
          node_pad[((size_t)m * MPAD + nr) * KPAD + nc] = (_Float16)v;
        }
      }
    }
}

// ---------------------------------------------------------------------------
// K2: fused GATv2 layer 1, one workgroup per graph (batch element).
// xl/xr = node @ Wl/Wr via 128 single-WMMA tile jobs (K=32), then edge
// attention (4 heads x 64), per-dst softmax via CSR, aggregation + ELU.
// ---------------------------------------------------------------------------
__global__ __launch_bounds__(256) void k_gat1(const _Float16* __restrict__ node_pad,
                                              const _Float16* __restrict__ wlT,
                                              const _Float16* __restrict__ wrT,
                                              const float* __restrict__ att1,
                                              const float* __restrict__ b1,
                                              const int* __restrict__ esrc,
                                              const int* __restrict__ edst,
                                              const int* __restrict__ coff,
                                              const int* __restrict__ ceid,
                                              _Float16* __restrict__ h1) {
  __shared__ float xl[N_NODES * GAT_HID];  // 49 KB
  __shared__ float xr[N_NODES * GAT_HID];  // 49 KB
  __shared__ float sc[E_CNT * HEADS];      // 5.7 KB
  const int b = blockIdx.x;
  const _Float16* nb = node_pad + (size_t)b * MPAD * KPAD;
  const int tid = threadIdx.x, wave = tid >> 5, lane = tid & 31;

  __builtin_prefetch(wlT, 0, 1);
  __builtin_prefetch(wrT, 0, 1);

  // 2 GEMMs x 4 row-tiles x 16 col-tiles = 128 jobs
  for (int job = wave; job < 128; job += 8) {
    int g = job >> 6, rt = (job >> 4) & 3, ct = job & 15;
    const _Float16* wT = g ? wrT : wlT;
    v16h a = gfrag(nb, rt * 16, KPAD, 0);
    v16h bm = gfrag(wT, ct * 16, KPAD, 0);
    v8f c = {};
    c = wmma16(a, bm, c);
    float* dst = g ? xr : xl;
    int col = ct * 16 + (lane & 15);
    for (int r = 0; r < 8; r++) {
      int row = rt * 16 + r + ((lane < 16) ? 0 : 8);
      if (row < N_NODES) dst[row * GAT_HID + col] = c[r];
    }
  }
  __syncthreads();

  // edge scores: (e, head) jobs
  for (int job = tid; job < E_CNT * HEADS; job += 256) {
    int e = job >> 2, h = job & 3;
    const float* pl = &xl[esrc[e] * GAT_HID + h * 64];
    const float* pr = &xr[edst[e] * GAT_HID + h * 64];
    const float* pa = &att1[h * 64];
    float acc = 0.f;
    for (int c0 = 0; c0 < 64; c0++) {
      float v = pl[c0] + pr[c0];
      v = (v > 0.f) ? v : 0.2f * v;
      acc += pa[c0] * v;
    }
    sc[e * HEADS + h] = acc;
  }
  __syncthreads();

  // softmax over in-edges per (node, head)
  for (int job = tid; job < N_NODES * HEADS; job += 256) {
    int n = job >> 2, h = job & 3;
    int s0 = coff[n], s1 = coff[n + 1];
    float m = -3.4e38f;
    for (int i = s0; i < s1; i++) m = fmaxf(m, sc[ceid[i] * HEADS + h]);
    float den = 0.f;
    for (int i = s0; i < s1; i++) den += __expf(sc[ceid[i] * HEADS + h] - m);
    float inv = 1.f / den;
    for (int i = s0; i < s1; i++) {
      int e = ceid[i];
      sc[e * HEADS + h] = __expf(sc[e * HEADS + h] - m) * inv;
    }
  }
  __syncthreads();

  // aggregate + bias + ELU -> h1 (f16)
  for (int job = tid; job < N_NODES * GAT_HID; job += 256) {
    int n = job >> 8, ch = job & 255, h = ch >> 6;
    int s0 = coff[n], s1 = coff[n + 1];
    float acc = 0.f;
    for (int i = s0; i < s1; i++) {
      int e = ceid[i];
      acc += sc[e * HEADS + h] * xl[esrc[e] * GAT_HID + ch];
    }
    acc += b1[ch];
    acc = (acc > 0.f) ? acc : (__expf(acc) - 1.f);
    h1[((size_t)b * N_NODES + n) * GAT_HID + ch] = (_Float16)acc;
  }
}

// ---------------------------------------------------------------------------
// K3: fused GATv2 layer 2 + mean-pool + FC, one workgroup per graph.
// h staged once into LDS (async, zero-padded to 64 rows); weight slabs
// (2 gemms x 16 cols x 256 K) double-buffered through LDS (async).
// Wave w -> (gemm = w>>2, rowtile = w&3); uniform ct loop => EXEC all-1 WMMA.
// ---------------------------------------------------------------------------
__global__ __launch_bounds__(256) void k_gat2(const _Float16* __restrict__ h1,
                                              const _Float16* __restrict__ wlT2,
                                              const _Float16* __restrict__ wrT2,
                                              const float* __restrict__ att2,
                                              const float* __restrict__ b2,
                                              const _Float16* __restrict__ node_pad,
                                              const float* __restrict__ fc_w,
                                              const float* __restrict__ fc_b,
                                              const int* __restrict__ esrc,
                                              const int* __restrict__ edst,
                                              const int* __restrict__ coff,
                                              const int* __restrict__ ceid,
                                              float* __restrict__ out) {
  __shared__ _Float16 sH[MPAD * GAT_HID];          // 32 KB (zero-padded rows)
  __shared__ _Float16 sW[2][2][16 * GAT_HID];      // 2 x 16 KB
  __shared__ _Float16 xl[N_NODES * GAT_HID];       // 24.5 KB
  __shared__ _Float16 xr[N_NODES * GAT_HID];       // 24.5 KB
  __shared__ _Float16 gbuf[N_NODES * GAT_HID];     // 24.5 KB
  __shared__ float sc[E_CNT];
  __shared__ float pooled[NODE_DIM + GAT_HID];
  const int b = blockIdx.x;
  const _Float16* hb = h1 + (size_t)b * N_NODES * GAT_HID;
  const int tid = threadIdx.x, wave = tid >> 5, lane = tid & 31;

  auto stageW = [&](int ct, int buf) {
    // 2 gemms x (16 x 256 halves = 512 x 16B chunks) = 1024 chunks
    for (int u = tid; u < 1024; u += 256) {
      int gg = u >> 9;
      int v = u & 511;
      int row = v >> 5, c8 = (v & 31) * 8;
      const _Float16* src = (gg ? wrT2 : wlT2) + (size_t)(ct * 16 + row) * GAT_HID + c8;
      async_g2l_b128((unsigned)(uintptr_t)&sW[buf][gg][row * GAT_HID + c8], src);
    }
  };

  // stage activations: rows < 49 async from global, rows >= 49 zeroed
  for (int u = tid; u < (N_NODES * GAT_HID) / 8; u += 256) {  // 1568 chunks
    int row = u >> 5, c8 = (u & 31) * 8;
    async_g2l_b128((unsigned)(uintptr_t)&sH[row * GAT_HID + c8],
                   hb + (size_t)row * GAT_HID + c8);
  }
  for (int u = tid; u < ((MPAD - N_NODES) * GAT_HID) / 8; u += 256) {  // 480 chunks
    int uu = u + (N_NODES * GAT_HID) / 8;
    int row = uu >> 5, c8 = (uu & 31) * 8;
    *reinterpret_cast<uint4*>(&sH[row * GAT_HID + c8]) = make_uint4(0u, 0u, 0u, 0u);
  }
  stageW(0, 0);
  wait_async0();
  __syncthreads();

  const int g = wave >> 2, rt = wave & 3;
  int cur = 0;
  for (int ct = 0; ct < 16; ct++) {
    if (ct + 1 < 16) stageW(ct + 1, cur ^ 1);   // overlap next slab fill
    v8f c = {};
    for (int kk = 0; kk < 8; kk++) {
      v16h a = sfrag(sH, rt * 16, GAT_HID, kk * 32);
      v16h bm = sfrag(sW[cur][g], 0, GAT_HID, kk * 32);
      c = wmma16(a, bm, c);
    }
    _Float16* dst = g ? xr : xl;
    int col = ct * 16 + (lane & 15);
    for (int r = 0; r < 8; r++) {
      int row = rt * 16 + r + ((lane < 16) ? 0 : 8);
      if (row < N_NODES) dst[row * GAT_HID + col] = (_Float16)c[r];
    }
    wait_async0();
    __syncthreads();
    cur ^= 1;
  }

  // edge scores (1 head, 256-dot)
  for (int e = tid; e < E_CNT; e += 256) {
    const _Float16* pl = &xl[esrc[e] * GAT_HID];
    const _Float16* pr = &xr[edst[e] * GAT_HID];
    float acc = 0.f;
    for (int c0 = 0; c0 < GAT_HID; c0++) {
      float v = (float)pl[c0] + (float)pr[c0];
      v = (v > 0.f) ? v : 0.2f * v;
      acc += att2[c0] * v;
    }
    sc[e] = acc;
  }
  __syncthreads();

  for (int n = tid; n < N_NODES; n += 256) {
    int s0 = coff[n], s1 = coff[n + 1];
    float m = -3.4e38f;
    for (int i = s0; i < s1; i++) m = fmaxf(m, sc[ceid[i]]);
    float den = 0.f;
    for (int i = s0; i < s1; i++) den += __expf(sc[ceid[i]] - m);
    float inv = 1.f / den;
    for (int i = s0; i < s1; i++) { int e = ceid[i]; sc[e] = __expf(sc[e] - m) * inv; }
  }
  __syncthreads();

  for (int job = tid; job < N_NODES * GAT_HID; job += 256) {
    int n = job >> 8, ch = job & 255;
    int s0 = coff[n], s1 = coff[n + 1];
    float acc = 0.f;
    for (int i = s0; i < s1; i++) {
      int e = ceid[i];
      acc += sc[e] * (float)xl[esrc[e] * GAT_HID + ch];
    }
    gbuf[n * GAT_HID + ch] = (_Float16)(acc + b2[ch]);
  }
  __syncthreads();

  // mean pooling: pooled = concat(node.mean(49), g.mean(49))
  for (int j = tid; j < NODE_DIM + GAT_HID; j += 256) {
    float acc = 0.f;
    if (j < NODE_DIM) {
      const _Float16* nb = node_pad + (size_t)b * MPAD * KPAD;
      for (int n = 0; n < N_NODES; n++) acc += (float)nb[n * KPAD + j];
    } else {
      int ch = j - NODE_DIM;
      for (int n = 0; n < N_NODES; n++) acc += (float)gbuf[n * GAT_HID + ch];
    }
    pooled[j] = acc * (1.f / (float)N_NODES);
  }
  __syncthreads();

  if (tid < NUM_CLS) {
    float acc = fc_b[tid];
    for (int j = 0; j < NODE_DIM + GAT_HID; j++)
      acc += pooled[j] * fc_w[j * NUM_CLS + tid];
    out[(size_t)b * NUM_CLS + tid] = acc;
  }
}

// ---------------------------------------------------------------------------
// Host: workspace layout + launches
// ---------------------------------------------------------------------------
static inline size_t alignup(size_t x) { return (x + 255) & ~(size_t)255; }

extern "C" void kernel_launch(void* const* d_in, const int* in_sizes, int n_in,
                              void* d_out, int out_size, void* d_ws, size_t ws_size,
                              hipStream_t stream) {
  const float* x     = (const float*)d_in[0];
  const float* tw    = (const float*)d_in[1];
  const float* tb    = (const float*)d_in[2];
  const float* g1wl  = (const float*)d_in[3];
  const float* g1wr  = (const float*)d_in[4];
  const float* g1att = (const float*)d_in[5];
  const float* g1b   = (const float*)d_in[6];
  const float* g2wl  = (const float*)d_in[7];
  const float* g2wr  = (const float*)d_in[8];
  const float* g2att = (const float*)d_in[9];
  const float* g2b   = (const float*)d_in[10];
  const float* fcw   = (const float*)d_in[11];
  const float* fcb   = (const float*)d_in[12];
  const int*   ei    = (const int*)d_in[13];
  float* outp = (float*)d_out;

  char* ws = (char*)d_ws;
  size_t off = 0;
  _Float16* xb = (_Float16*)(ws + off);      off = alignup(off + (size_t)BATCH * FEAT_DIM * 2);
  _Float16* wtT = (_Float16*)(ws + off);     off = alignup(off + (size_t)NPAD_TR * FEAT_DIM * 2);
  _Float16* node_pad = (_Float16*)(ws + off);off = alignup(off + (size_t)BATCH * MPAD * KPAD * 2);
  _Float16* wlT1 = (_Float16*)(ws + off);    off = alignup(off + (size_t)GAT_HID * KPAD * 2);
  _Float16* wrT1 = (_Float16*)(ws + off);    off = alignup(off + (size_t)GAT_HID * KPAD * 2);
  _Float16* wlT2 = (_Float16*)(ws + off);    off = alignup(off + (size_t)GAT_HID * GAT_HID * 2);
  _Float16* wrT2 = (_Float16*)(ws + off);    off = alignup(off + (size_t)GAT_HID * GAT_HID * 2);
  _Float16* h1 = (_Float16*)(ws + off);      off = alignup(off + (size_t)BATCH * N_NODES * GAT_HID * 2);
  int* esrc = (int*)(ws + off);              off = alignup(off + E_CNT * 4);
  int* edst = (int*)(ws + off);              off = alignup(off + E_CNT * 4);
  int* coff = (int*)(ws + off);              off = alignup(off + (N_NODES + 1) * 4);
  int* ceid = (int*)(ws + off);              off = alignup(off + E_CNT * 4);
  (void)ws_size; (void)n_in; (void)in_sizes; (void)out_size;

  // --- K0: pack / convert / CSR ---
  {
    size_t n8 = (size_t)BATCH * MPAD * KPAD / 8;
    k_zero_f16<<<(unsigned)((n8 + 255) / 256), 256, 0, stream>>>(node_pad, n8);
  }
  {
    size_t n = (size_t)BATCH * FEAT_DIM;
    k_cvt_f16<<<(unsigned)((n + 255) / 256), 256, 0, stream>>>(x, xb, n);
  }
  {
    size_t n = (size_t)NPAD_TR * FEAT_DIM;
    k_pack_wt<<<(unsigned)((n + 255) / 256), 256, 0, stream>>>(tw, wtT);
  }
  k_pack_g1<<<(GAT_HID * KPAD + 255) / 256, 256, 0, stream>>>(g1wl, g1wr, wlT1, wrT1);
  k_pack_g2<<<(GAT_HID * GAT_HID + 255) / 256, 256, 0, stream>>>(g2wl, g2wr, wlT2, wrT2);
  k_build_csr<<<1, 32, 0, stream>>>(ei, esrc, edst, coff, ceid);

  // --- K1: transform GEMM ---
  {
    dim3 grid(NPAD_TR / 64, BATCH / 128);
    k_transform<<<grid, 256, 0, stream>>>(xb, wtT, tb, node_pad);
  }

  // --- K2: GAT layer 1 (fused per graph) ---
  k_gat1<<<BATCH, 256, 0, stream>>>(node_pad, wlT1, wrT1, g1att, g1b,
                                    esrc, edst, coff, ceid, h1);

  // --- K3: GAT layer 2 + pooling + FC ---
  k_gat2<<<BATCH, 256, 0, stream>>>(h1, wlT2, wrT2, g2att, g2b, node_pad,
                                    fcw, fcb, esrc, edst, coff, ceid, outp);
}
